// GPSA_36610301231709
// MI455X (gfx1250) — matrix operand order repaired and checked by reading, not compiled, not run on hardware
//
#include <hip/hip_runtime.h>
#include <hip/hip_bf16.h>

// ---------------------------------------------------------------------------
// GPSA (gated positional self-attention), fused flash-style for gfx1250.
//   B=4, N=1024, C=256, H=8, hd=32.  All GEMMs on v_wmma_f32_16x16x32_f16.
//   Attention K/V tiles staged to LDS with global_load_async_to_lds_b128
//   (double-buffered, block-cooperative), tracked with ASYNCcnt.
// ---------------------------------------------------------------------------

#define B_ 4
#define N_ 1024
#define C_ 256
#define H_ 8
#define HD_ 32

typedef _Float16 half_t;
typedef __attribute__((ext_vector_type(16))) _Float16 v16h;
typedef __attribute__((ext_vector_type(8)))  _Float16 v8h;
typedef __attribute__((ext_vector_type(4)))  _Float16 v4h;
typedef __attribute__((ext_vector_type(8)))  float    v8f;
typedef __attribute__((ext_vector_type(4)))  float    v4f;

union V16U { v16h v; v8h h[2]; };

__device__ __forceinline__ v8f zero8() {
  v8f z;
#pragma unroll
  for (int j = 0; j < 8; ++j) z[j] = 0.0f;
  return z;
}

__device__ __forceinline__ v8f wmma_f16(v16h a, v16h b, v8f c) {
  return __builtin_amdgcn_wmma_f32_16x16x32_f16(false, a, false, b, (short)0, c,
                                                false, false);
}

// A-matrix fragment (16x32 f16, M x K) from an f16 row-major tile (ld halves).
__device__ __forceinline__ v16h load_a_frag_f16(const half_t* tile, int ld,
                                                int lane) {
  int r  = lane & 15;
  int kb = (lane >> 4) << 4;
  const half_t* p = tile + (size_t)r * ld + kb;
  V16U u;
  u.h[0] = *(const v8h*)(p);
  u.h[1] = *(const v8h*)(p + 8);
  return u.v;
}

// B-matrix fragment (32x16 f16, K x N): pointer to this lane's 16 contiguous
// halves (column n=lane&15, K-range selected by lane>=16).
__device__ __forceinline__ v16h load_b_frag_f16(const half_t* p) {
  V16U u;
  u.h[0] = *(const v8h*)(p);
  u.h[1] = *(const v8h*)(p + 8);
  return u.v;
}

// ---- CDNA5 async global->LDS copy (ASYNCcnt-tracked) ----------------------
// ISA §15.18: GLOBAL_LOAD_ASYNC_TO_LDS_B128  vdst(LDS addr), vaddr(64b), off
// Generic pointers to __shared__ carry the LDS offset in their low 32 bits.
__device__ __forceinline__ void async_ld16(half_t* lds_dst, const half_t* gsrc) {
  uint32_t lo = (uint32_t)(uintptr_t)lds_dst;
  uint64_t ga = (uint64_t)(uintptr_t)gsrc;
  asm volatile("global_load_async_to_lds_b128 %0, %1, off"
               :
               : "v"(lo), "v"(ga)
               : "memory");
}
__device__ __forceinline__ void wait_async0() {
  asm volatile("s_wait_asynccnt 0x0" ::: "memory");
}

// ---------------------------------------------------------------------------
// Kernel 0: vectorized f32 -> f16 conversion (4 elements / thread).
// ---------------------------------------------------------------------------
__global__ __launch_bounds__(256) void cvt4_kernel(const float* __restrict__ s,
                                                   half_t* __restrict__ d,
                                                   int n4) {
  int i = blockIdx.x * 256 + threadIdx.x;
  if (i < n4) {
    v4f f = *(const v4f*)(s + (size_t)4 * i);
    v4h h;
#pragma unroll
    for (int j = 0; j < 4; ++j) h[j] = (_Float16)f[j];
    *(v4h*)(d + (size_t)4 * i) = h;
  }
}

// ---------------------------------------------------------------------------
// Kernel 1: fused QKV projection (all-f16 operands).
//   w16 = [Wqk (512 rows) ; Wv (256 rows)], row-major, K=256.
//   q[bh][n][d], k[bh][n][d]  f16;  vT[bh][d][n]  f16 (transposed for PV).
// ---------------------------------------------------------------------------
__global__ __launch_bounds__(256) void gpsa_qkv_kernel(
    const half_t* __restrict__ x16, const half_t* __restrict__ w16,
    half_t* __restrict__ q16, half_t* __restrict__ k16,
    half_t* __restrict__ vT16) {
  int lane  = threadIdx.x & 31;
  int wave  = threadIdx.x >> 5;
  int mtile = blockIdx.x;  // 0..255
  const half_t* xt = x16 + (size_t)mtile * 16 * C_;

  v8f acc[6];
#pragma unroll
  for (int t = 0; t < 6; ++t) acc[t] = zero8();

  int nloc = lane & 15;
  int kb2  = (lane >> 4) * 16;

  for (int kc = 0; kc < C_; kc += 32) {
    v16h a = load_a_frag_f16(xt + kc, C_, lane);
    v16h bf[6];
#pragma unroll
    for (int t = 0; t < 6; ++t) {
      int n = (wave + t * 8) * 16 + nloc;
      bf[t] = load_b_frag_f16(w16 + (size_t)n * C_ + kc + kb2);
    }
#pragma unroll
    for (int t = 0; t < 6; ++t) acc[t] = wmma_f16(a, bf[t], acc[t]);
  }

  int rb   = (lane >> 4) * 8;
  int row0 = mtile * 16;
  int b    = row0 >> 10;
  int nnb  = (row0 & 1023);

#pragma unroll
  for (int t = 0; t < 6; ++t) {
    int ct = wave + t * 8;
    int n  = ct * 16 + nloc;
    if (n < 512) {
      int s = n >> 8, h = (n >> 5) & 7, d = n & 31;
      half_t* dst = (s == 0 ? q16 : k16) + (size_t)(b * H_ + h) * N_ * HD_;
#pragma unroll
      for (int j = 0; j < 8; ++j)
        dst[(size_t)(nnb + rb + j) * HD_ + d] = (half_t)acc[t][j];
    } else {
      int h = ((n - 512) >> 5) & 7, d = n & 31;
      half_t* dst = vT16 + ((size_t)(b * H_ + h) * HD_ + d) * N_ + nnb + rb;
      v8h pk;
#pragma unroll
      for (int j = 0; j < 8; ++j) pk[j] = (half_t)acc[t][j];
      *(v8h*)dst = pk;
    }
  }
}

// ---------------------------------------------------------------------------
// Online-softmax step for one stream over a 32-key tile.
// ---------------------------------------------------------------------------
__device__ __forceinline__ void online_step(v8f s0, v8f s1, v8f& m, v8f& l,
                                            v8f& o0, v8f& o1, half_t* lds,
                                            int lane, v16h vf0, v16h vf1) {
  v8f tmax;
#pragma unroll
  for (int j = 0; j < 8; ++j) tmax[j] = fmaxf(s0[j], s1[j]);
#pragma unroll
  for (int off = 1; off < 16; off <<= 1) {
#pragma unroll
    for (int j = 0; j < 8; ++j)
      tmax[j] = fmaxf(tmax[j], __shfl_xor(tmax[j], off, 32));
  }

  v8f alpha, p0, p1, tsum;
#pragma unroll
  for (int j = 0; j < 8; ++j) {
    float mn = fmaxf(m[j], tmax[j]);
    alpha[j] = __expf(m[j] - mn);
    p0[j]    = __expf(s0[j] - mn);
    p1[j]    = __expf(s1[j] - mn);
    tsum[j]  = p0[j] + p1[j];
    m[j]     = mn;
  }
#pragma unroll
  for (int off = 1; off < 16; off <<= 1) {
#pragma unroll
    for (int j = 0; j < 8; ++j) tsum[j] += __shfl_xor(tsum[j], off, 32);
  }
#pragma unroll
  for (int j = 0; j < 8; ++j) {
    l[j]  = l[j] * alpha[j] + tsum[j];
    o0[j] *= alpha[j];
    o1[j] *= alpha[j];
  }

  // P (C/D layout) -> per-wave LDS 16x32 -> reload in A layout.
  int rb = (lane >> 4) * 8, c = lane & 15;
#pragma unroll
  for (int j = 0; j < 8; ++j) {
    lds[(rb + j) * 32 + c]      = (half_t)p0[j];
    lds[(rb + j) * 32 + c + 16] = (half_t)p1[j];
  }
  __builtin_amdgcn_wave_barrier();  // DS ops are in-order within a wave
  v16h pa = load_a_frag_f16(lds, 32, lane);

  o0 = wmma_f16(pa, vf0, o0);
  o1 = wmma_f16(pa, vf1, o1);
}

// ---------------------------------------------------------------------------
// Kernel 2: fused dual-stream attention.
// Grid (N/64, H, B), block 128 (4 waves); each wave owns 16 query rows.
// K/V tiles double-buffered in LDS via async copies shared by all 4 waves.
// ---------------------------------------------------------------------------
__global__ __launch_bounds__(128) void gpsa_attn_kernel(
    const half_t* __restrict__ q16, const half_t* __restrict__ k16,
    const half_t* __restrict__ vT16, const float* __restrict__ coord,
    const float* __restrict__ Wpos, const float* __restrict__ bpos,
    const float* __restrict__ gating, half_t* __restrict__ ctx16) {
  __shared__ alignas(16) half_t ptile[4][16 * 32];  // per-wave P transpose
  __shared__ alignas(16) half_t kbuf[2][32 * 32];   // keys x d
  __shared__ alignas(16) half_t vbuf[2][32 * 32];   // d x keys

  int tid   = threadIdx.x;
  int lane  = tid & 31;
  int wave  = tid >> 5;
  int b     = blockIdx.z;
  int h     = blockIdx.y;
  int qbase = blockIdx.x * 64 + wave * 16;
  int bh    = b * H_ + h;

  const half_t* kbase = k16 + (size_t)(bh * N_) * HD_;
  const half_t* vbase = vT16 + (size_t)bh * HD_ * N_;

  // Each of 128 threads async-copies one 16B chunk per matrix per tile.
  int trow = tid >> 2;           // 0..31 (key index for K, d index for V)
  int tc   = (tid & 3) * 8;      // half offset within 64B row

  // prologue: stage tile 0 into buffer 0
  async_ld16(&kbuf[0][trow * 32 + tc], kbase + (size_t)trow * HD_ + tc);
  async_ld16(&vbuf[0][trow * 32 + tc], vbase + (size_t)trow * N_ + tc);

  const half_t* qb = q16 + (size_t)(bh * N_ + qbase) * HD_;
  v16h qa = load_a_frag_f16(qb, HD_, lane);

  float w0 = Wpos[h * 4 + 0], w1 = Wpos[h * 4 + 1];
  float w2 = Wpos[h * 4 + 2], w3 = Wpos[h * 4 + 3];
  float bp = bpos[h];
  float g  = 1.0f / (1.0f + __expf(-gating[h]));
  const float scale = 0.17677669529663687f;  // 1/sqrt(32)

  int rbase = (lane >> 4) * 8;
  float qcx[8], qcy[8], qcz[8];
#pragma unroll
  for (int j = 0; j < 8; ++j) {
    const float* c = coord + ((size_t)b * N_ + qbase + rbase + j) * 3;
    qcx[j] = c[0]; qcy[j] = c[1]; qcz[j] = c[2];
  }

  v8f mP, lP, mQ, lQ;
  v8f oP0 = zero8(), oP1 = zero8(), oQ0 = zero8(), oQ1 = zero8();
#pragma unroll
  for (int j = 0; j < 8; ++j) {
    mP[j] = -1e30f; mQ[j] = -1e30f; lP[j] = 0.0f; lQ[j] = 0.0f;
  }

  int col0 = lane & 15;
  int kb2  = (lane >> 4) * 16;
  half_t* lds = ptile[wave];

  wait_async0();
  __syncthreads();

  for (int m0 = 0; m0 < N_; m0 += 32) {
    int cb = (m0 >> 5) & 1;
    int nb = cb ^ 1;
    int mn = (m0 + 32 < N_) ? (m0 + 32) : 0;  // branchless wrap on last iter

    // stage next tile (overlaps with compute below)
    async_ld16(&kbuf[nb][trow * 32 + tc], kbase + (size_t)(mn + trow) * HD_ + tc);
    async_ld16(&vbuf[nb][trow * 32 + tc], vbase + (size_t)trow * N_ + mn + tc);

    // fragments from LDS (ds_load_b128 pairs)
    v16h kf0 = load_b_frag_f16(&kbuf[cb][(size_t)col0 * 32 + kb2]);
    v16h kf1 = load_b_frag_f16(&kbuf[cb][(size_t)(col0 + 16) * 32 + kb2]);
    v16h vf0 = load_b_frag_f16(&vbuf[cb][(size_t)col0 * 32 + kb2]);
    v16h vf1 = load_b_frag_f16(&vbuf[cb][(size_t)(col0 + 16) * 32 + kb2]);

    // ---- patch logits
    v8f s0 = wmma_f16(qa, kf0, zero8());
    v8f s1 = wmma_f16(qa, kf1, zero8());
#pragma unroll
    for (int j = 0; j < 8; ++j) { s0[j] *= scale; s1[j] *= scale; }

    // ---- positional logits (VALU)
    const float* ck0 = coord + ((size_t)b * N_ + m0 + col0) * 3;
    const float* ck1 = ck0 + 16 * 3;
    float k0x = ck0[0], k0y = ck0[1], k0z = ck0[2];
    float k1x = ck1[0], k1y = ck1[1], k1z = ck1[2];
    v8f t0, t1;
#pragma unroll
    for (int j = 0; j < 8; ++j) {
      float dx = qcx[j] - k0x, dy = qcy[j] - k0y, dz = qcz[j] - k0z;
      float dd = __fsqrt_rn(dx * dx + dy * dy + dz * dz);
      t0[j] = w0 * dx + w1 * dy + w2 * dz + w3 * dd + bp;
      dx = qcx[j] - k1x; dy = qcy[j] - k1y; dz = qcz[j] - k1z;
      dd = __fsqrt_rn(dx * dx + dy * dy + dz * dz);
      t1[j] = w0 * dx + w1 * dy + w2 * dz + w3 * dd + bp;
    }

    // ---- online softmax + PV, both streams (hides async latency)
    online_step(s0, s1, mP, lP, oP0, oP1, lds, lane, vf0, vf1);  // patch
    online_step(t0, t1, mQ, lQ, oQ0, oQ1, lds, lane, vf0, vf1);  // positional

    wait_async0();    // my chunks of the next tile have landed
    __syncthreads();  // everyone's chunks landed; all reads of cb done
  }

  // Combine: out = (1-g)*O_patch/l_patch + g*O_pos/l_pos
  // (combined attention rows sum to (1-g)+g = 1 -> no extra renorm)
  half_t* cbp = ctx16 + ((size_t)b * N_ + qbase) * C_ + h * HD_;
#pragma unroll
  for (int j = 0; j < 8; ++j) {
    float ip = (1.0f - g) / lP[j];
    float iq = g / lQ[j];
    size_t ro = (size_t)(rbase + j) * C_;
    cbp[ro + col0]      = (half_t)(oP0[j] * ip + oQ0[j] * iq);
    cbp[ro + col0 + 16] = (half_t)(oP1[j] * ip + oQ1[j] * iq);
  }
}

// ---------------------------------------------------------------------------
// Kernel 3: output projection  out = ctx @ Wproj^T + bproj   (f32 out)
// ---------------------------------------------------------------------------
__global__ __launch_bounds__(256) void gpsa_proj_kernel(
    const half_t* __restrict__ ctx16, const half_t* __restrict__ wp16,
    const float* __restrict__ bproj, float* __restrict__ out) {
  int lane  = threadIdx.x & 31;
  int wave  = threadIdx.x >> 5;
  int mtile = blockIdx.x;
  const half_t* ct = ctx16 + (size_t)mtile * 16 * C_;

  v8f acc[2];
#pragma unroll
  for (int t = 0; t < 2; ++t) acc[t] = zero8();

  int nloc = lane & 15;
  int kb2  = (lane >> 4) * 16;

  for (int kc = 0; kc < C_; kc += 32) {
    v16h a = load_a_frag_f16(ct + kc, C_, lane);
    v16h bf[2];
#pragma unroll
    for (int t = 0; t < 2; ++t) {
      int n = (wave + t * 8) * 16 + nloc;
      bf[t] = load_b_frag_f16(wp16 + (size_t)n * C_ + kc + kb2);
    }
#pragma unroll
    for (int t = 0; t < 2; ++t) acc[t] = wmma_f16(a, bf[t], acc[t]);
  }

  int rb = (lane >> 4) * 8;
#pragma unroll
  for (int t = 0; t < 2; ++t) {
    int n    = (wave + t * 8) * 16 + nloc;
    float bb = bproj[n];
#pragma unroll
    for (int j = 0; j < 8; ++j)
      out[(size_t)(mtile * 16 + rb + j) * C_ + n] = acc[t][j] + bb;
  }
}

// ---------------------------------------------------------------------------
extern "C" void kernel_launch(void* const* d_in, const int* in_sizes, int n_in,
                              void* d_out, int out_size, void* d_ws,
                              size_t ws_size, hipStream_t stream) {
  const float* x      = (const float*)d_in[0];  // (B,N,C)
  const float* coord  = (const float*)d_in[1];  // (B,N,3)
  const float* Wqk    = (const float*)d_in[2];  // (2C,C)
  const float* Wv     = (const float*)d_in[3];  // (C,C)
  const float* Wpos   = (const float*)d_in[4];  // (H,4)
  const float* bpos   = (const float*)d_in[5];  // (H)
  const float* Wproj  = (const float*)d_in[6];  // (C,C)
  const float* bproj  = (const float*)d_in[7];  // (C)
  const float* gating = (const float*)d_in[8];  // (H)
  float* out = (float*)d_out;

  const size_t xels  = (size_t)B_ * N_ * C_;        // 1,048,576
  const size_t qels  = (size_t)B_ * H_ * N_ * HD_;  // 1,048,576
  const size_t wqkEl = (size_t)2 * C_ * C_;         // 131,072
  const size_t wvEl  = (size_t)C_ * C_;             // 65,536

  half_t* x16   = (half_t*)d_ws;
  half_t* w16   = x16 + xels;           // [Wqk ; Wv] packed, 768 x 256
  half_t* wp16  = w16 + wqkEl + wvEl;   // Wproj, 256 x 256
  half_t* q16   = wp16 + wvEl;
  half_t* k16   = q16 + qels;
  half_t* vT16  = k16 + qels;
  half_t* ctx16 = vT16 + qels;

  cvt4_kernel<<<dim3((int)(xels / 4 / 256)), dim3(256), 0, stream>>>(
      x, x16, (int)(xels / 4));
  cvt4_kernel<<<dim3((int)(wqkEl / 4 / 256)), dim3(256), 0, stream>>>(
      Wqk, w16, (int)(wqkEl / 4));
  cvt4_kernel<<<dim3((int)(wvEl / 4 / 256)), dim3(256), 0, stream>>>(
      Wv, w16 + wqkEl, (int)(wvEl / 4));
  cvt4_kernel<<<dim3((int)(wvEl / 4 / 256)), dim3(256), 0, stream>>>(
      Wproj, wp16, (int)(wvEl / 4));

  gpsa_qkv_kernel<<<dim3((B_ * N_) / 16), dim3(256), 0, stream>>>(
      x16, w16, q16, k16, vT16);

  gpsa_attn_kernel<<<dim3(N_ / 64, H_, B_), dim3(128), 0, stream>>>(
      q16, k16, vT16, coord, Wpos, bpos, gating, ctx16);

  gpsa_proj_kernel<<<dim3((B_ * N_) / 16), dim3(256), 0, stream>>>(
      ctx16, wp16, bproj, out);
}
